// model_bbox_encoder_73950746902690
// MI455X (gfx1250) — compile-verified, matrix operand order checked by reading, not run on hardware
//
#include <hip/hip_runtime.h>

typedef _Float16 half_t;
typedef _Float16 v16h __attribute__((ext_vector_type(16)));
typedef _Float16 v8h  __attribute__((ext_vector_type(8)));
typedef float    v8f  __attribute__((ext_vector_type(8)));
typedef int      v4i  __attribute__((ext_vector_type(4)));

#define B_TOT   4096
#define T_LEN   64
#define H_D     512
#define G3      1536
#define BTILE   16       // batch rows per workgroup -> 256 workgroups
#define THREADS 256      // 8 waves (wave32)
#define NWAVES  8
#define JT      4        // 16-col tiles per gate per wave (4*16 = 64 cols/gate)
#define STAGE_HALVES (192 * 32)   // per wave per buffer: 192 B-rows x 32 halves = 12 KB

union ABfrag { v16h v; v8h h[2]; };

// ---------- CDNA5 async global->LDS copy (ASYNCcnt-tracked) ----------
#if __has_builtin(__builtin_amdgcn_global_load_async_to_lds_b128)
#define GRU_ASYNC_BUILTIN 1
#endif

typedef __attribute__((address_space(1))) v4i* gptr_b128;
typedef __attribute__((address_space(3))) v4i* lptr_b128;

__device__ __forceinline__ void async_copy_b128(const half_t* g, half_t* l) {
#ifdef GRU_ASYNC_BUILTIN
    __builtin_amdgcn_global_load_async_to_lds_b128(
        (gptr_b128)(g), (lptr_b128)(l), 0, 0);
#else
    unsigned loff = (unsigned)(__UINTPTR_TYPE__)(__attribute__((address_space(3))) half_t*)(l);
    asm volatile("global_load_async_to_lds_b128 %0, %1, off"
                 :: "v"(loff), "v"(g) : "memory");
#endif
}

template <int N>
__device__ __forceinline__ void wait_async() {
#if __has_builtin(__builtin_amdgcn_s_wait_asynccnt)
    __builtin_amdgcn_s_wait_asynccnt((unsigned short)N);
#else
    asm volatile("s_wait_asynccnt %0" :: "n"(N) : "memory");
#endif
}

// ---- prep: convert w_hh [1536,512] f32 -> f16 in workspace (L2-resident) ----
__global__ void gru_cvt_whh_f16(const float* __restrict__ w,
                                half_t* __restrict__ o, int n) {
    int i = blockIdx.x * blockDim.x + threadIdx.x;
    if (i < n) o[i] = (half_t)w[i];
}

__global__ __launch_bounds__(THREADS)
void gru_scan_kernel(const float*  __restrict__ x,      // [B, T, 4]
                     const float*  __restrict__ h0,     // [B, 512]
                     const float*  __restrict__ w_ih,   // [1536, 4]
                     const float*  __restrict__ b_ih,   // [1536]
                     const float*  __restrict__ b_hh,   // [1536]
                     const half_t* __restrict__ whh16,  // [1536, 512] f16
                     float*        __restrict__ out)    // [B, 512]
{
    __shared__ __align__(16) half_t h_s[BTILE][H_D];                  // 16 KB state
    __shared__ __align__(16) half_t bstage[2][NWAVES][STAGE_HALVES];  // 192 KB dbl-buf B
    __shared__ __align__(16) float4 wih_s[G3];                        // 24 KB
    __shared__ float bih_s[G3];                                       // 6 KB
    __shared__ float bhh_s[G3];                                       // 6 KB
    __shared__ __align__(16) float4 x_s[BTILE];                       // 256 B

    const int tid  = threadIdx.x;
    const int lane = tid & 31;
    const int wave = tid >> 5;
    const int b0   = blockIdx.x * BTILE;

    // ---- one-time preload ----
    for (int i = tid; i < G3; i += THREADS) {
        wih_s[i] = ((const float4*)w_ih)[i];
        bih_s[i] = b_ih[i];
        bhh_s[i] = b_hh[i];
    }
    for (int i = tid; i < BTILE * H_D; i += THREADS) {
        int m = i >> 9, j = i & (H_D - 1);
        h_s[m][j] = (half_t)h0[(size_t)(b0 + m) * H_D + j];
    }
    __syncthreads();

    const int hi = (lane >= 16) ? 1 : 0;   // K-half select per 16-bit WMMA layout
    const int lc = lane & 15;

    // per-lane staging geometry: 24 b128 async ops move 192 rows x 64B per wave
    const int lr  = lane >> 2;             // row within an 8-row group
    const int lco = (lane & 3) * 8;        // halves within a row (16B chunk)
    const half_t* gbase = whh16 + (size_t)(wave * 64 + lr) * H_D + lco;

    // prologue: stage k-chunk 0 into buffer 0
    {
        half_t* lp = &bstage[0][wave][0] + lr * 32 + lco;
        #pragma unroll
        for (int g = 0; g < 3; ++g)
            #pragma unroll
            for (int ii = 0; ii < 8; ++ii)
                async_copy_b128(gbase + (size_t)(g * 512 + ii * 8) * H_D,
                                lp + (g * 64 + ii * 8) * 32);
    }

    #pragma unroll 1
    for (int t = 0; t < T_LEN; ++t) {
        if (tid < BTILE)
            x_s[tid] = ((const float4*)x)[(size_t)(b0 + tid) * T_LEN + t];

        // acc[gate][jt]: same lane/VGPR across gates -> register-resident combine
        v8f acc[3][JT];
        #pragma unroll
        for (int g = 0; g < 3; ++g)
            #pragma unroll
            for (int jt = 0; jt < JT; ++jt)
                acc[g][jt] = (v8f)(0.0f);

        #pragma unroll 2
        for (int ks = 0; ks < 16; ++ks) {
            // prefetch next k-chunk (ks==15 wraps to kk=0: next timestep's first
            // chunk flies during the gate-combine phase below)
            {
                const int kn = ((ks + 1) & 15) * 32;
                half_t* lp = &bstage[(ks + 1) & 1][wave][0] + lr * 32 + lco;
                const half_t* gp = gbase + kn;
                #pragma unroll
                for (int g = 0; g < 3; ++g)
                    #pragma unroll
                    for (int ii = 0; ii < 8; ++ii)
                        async_copy_b128(gp + (size_t)(g * 512 + ii * 8) * H_D,
                                        lp + (g * 64 + ii * 8) * 32);
            }
            wait_async<24>();   // current buffer complete; next 24 still in flight

            const int kb = ks * 32 + hi * 8;
            ABfrag a;
            a.h[0] = *(const v8h*)&h_s[lc][kb];
            a.h[1] = *(const v8h*)&h_s[lc][kb + 16];

            const half_t* cur = &bstage[ks & 1][wave][0];
            #pragma unroll
            for (int g = 0; g < 3; ++g) {
                #pragma unroll
                for (int jt = 0; jt < JT; ++jt) {
                    ABfrag bf;
                    const half_t* bp = cur + (g * 64 + jt * 16 + lc) * 32 + hi * 8;
                    bf.h[0] = *(const v8h*)bp;
                    bf.h[1] = *(const v8h*)(bp + 16);
                    acc[g][jt] = __builtin_amdgcn_wmma_f32_16x16x32_f16(
                        false, a.v, false, bf.v, (short)0, acc[g][jt], false, false);
                }
            }
        }
        __syncthreads();   // all waves done reading h_s before it is rewritten

        // ---- fused gates, register-resident (C layout: VGPR r -> M=r+8*hi, N=lc) ----
        float4 xv[8];
        #pragma unroll
        for (int r = 0; r < 8; ++r)
            xv[r] = x_s[r + hi * 8];

        #pragma unroll
        for (int jt = 0; jt < JT; ++jt) {
            const int j = wave * 64 + jt * 16 + lc;
            const float4 wr = wih_s[j];
            const float4 wz = wih_s[j + 512];
            const float4 wn = wih_s[j + 1024];
            const float br = bih_s[j], bz = bih_s[j + 512], bn = bih_s[j + 1024];
            const float cr = bhh_s[j], cz = bhh_s[j + 512], cn = bhh_s[j + 1024];
            #pragma unroll
            for (int r = 0; r < 8; ++r) {
                const int m = r + hi * 8;
                const float4 q = xv[r];
                const float ir  = fmaf(q.x, wr.x, fmaf(q.y, wr.y, fmaf(q.z, wr.z, fmaf(q.w, wr.w, br))));
                const float iz  = fmaf(q.x, wz.x, fmaf(q.y, wz.y, fmaf(q.z, wz.z, fmaf(q.w, wz.w, bz))));
                const float inn = fmaf(q.x, wn.x, fmaf(q.y, wn.y, fmaf(q.z, wn.z, fmaf(q.w, wn.w, bn))));
                const float hr = acc[0][jt][r] + cr;
                const float hz = acc[1][jt][r] + cz;
                const float hn = acc[2][jt][r] + cn;
                const float rg = 1.0f / (1.0f + __expf(-(ir + hr)));
                const float zg = 1.0f / (1.0f + __expf(-(iz + hz)));
                const float av = inn + rg * hn;
                const float e2 = __expf(-2.0f * av);
                const float ng = (1.0f - e2) / (1.0f + e2);
                const float hold = (float)h_s[m][j];
                const float hnew = (1.0f - zg) * ng + zg * hold;
                h_s[m][j] = (half_t)hnew;
                if (t == T_LEN - 1)
                    out[(size_t)(b0 + m) * H_D + j] = hnew;
            }
        }
        __syncthreads();   // h_s updated before next step's A-fragment reads
    }
}

extern "C" void kernel_launch(void* const* d_in, const int* in_sizes, int n_in,
                              void* d_out, int out_size, void* d_ws, size_t ws_size,
                              hipStream_t stream) {
    const float* x    = (const float*)d_in[0];
    const float* h0   = (const float*)d_in[1];
    const float* w_ih = (const float*)d_in[2];
    const float* w_hh = (const float*)d_in[3];
    const float* b_ih = (const float*)d_in[4];
    const float* b_hh = (const float*)d_in[5];
    float* out = (float*)d_out;

    half_t* whh16 = (half_t*)d_ws;             // 1536*512*2 = 1.5 MB scratch
    const int nW = G3 * H_D;
    gru_cvt_whh_f16<<<(nW + 255) / 256, 256, 0, stream>>>(w_hh, whh16, nW);

    gru_scan_kernel<<<B_TOT / BTILE, THREADS, 0, stream>>>(
        x, h0, w_ih, b_ih, b_hh, whh16, out);
}
